// Critic_CrossAttentionMLP_3616362463944
// MI455X (gfx1250) — matrix operand order repaired and checked by reading, not compiled, not run on hardware
//
#include <hip/hip_runtime.h>

typedef _Float16 v8h  __attribute__((ext_vector_type(8)));
typedef _Float16 v16h __attribute__((ext_vector_type(16)));
typedef float    v8f  __attribute__((ext_vector_type(8)));

constexpr int kH   = 128;
constexpr int kS   = 64;
constexpr int kFOD = 15;
constexpr int kB   = 4096;
constexpr int kN   = kB * kS;   // 262144 tokens

#define WMMA(a, b, c) __builtin_amdgcn_wmma_f32_16x16x32_f16(false, (a), false, (b), (short)0, (c), false, false)

__device__ __forceinline__ v16h frag_from_ptrs(const _Float16* p0, const _Float16* p1) {
  v8h lo = *(const v8h*)p0;
  v8h hi = *(const v8h*)p1;
  v16h r;
#pragma unroll
  for (int i = 0; i < 8; ++i) { r[i] = lo[i]; r[i + 8] = hi[i]; }
  return r;
}

// B fragment (32x16 f16) from pre-swizzled global buffer: tile = kc*NT + nt
__device__ __forceinline__ v16h load_bfrag(const _Float16* __restrict__ base, int tile, int lane) {
  const _Float16* p = base + ((size_t)tile * 32 + lane) * 16;
  return frag_from_ptrs(p, p + 8);
}

// A fragment (16x32 f16) from row-major [16][stride] f16 LDS tile at K-chunk base kb.
// Lanes 0-15: row=lane, K={kb..kb+7, kb+16..kb+23}; lanes 16-31: row=lane-16, K offset +8.
__device__ __forceinline__ v16h load_afrag(const _Float16* tile, int stride, int kb, int lane) {
  int r   = lane & 15;
  int off = (lane >> 4) * 8;
  const _Float16* p = tile + r * stride + kb + off;
  return frag_from_ptrs(p, p + 16);
}

// ---------------------------------------------------------------------------
// Kernel 0: weights (row-major [Nout][Ksrc] f32) -> f16 B-fragment order,
// zero-padding K to a multiple of 32 for the small encoders.
// ---------------------------------------------------------------------------
__global__ void prep_frags(const float* __restrict__ wih, const float* __restrict__ whh,
                           const float* __restrict__ ain, const float* __restrict__ aout,
                           const float* __restrict__ m0,  const float* __restrict__ m1,
                           const float* __restrict__ eaw, const float* __restrict__ emw,
                           _Float16* __restrict__ fwih, _Float16* __restrict__ fwhh,
                           _Float16* __restrict__ fain, _Float16* __restrict__ faout,
                           _Float16* __restrict__ fm0,  _Float16* __restrict__ fm1,
                           _Float16* __restrict__ feaw, _Float16* __restrict__ femw) {
  const int seg_sz[8] = {49152, 49152, 49152, 16384, 16384, 16384, 4096, 4096};
  const int seg_nt[8] = {24, 24, 24, 8, 8, 8, 8, 8};
  const int seg_k [8] = {128, 128, 128, 128, 128, 128, 15, 4};
  const float* src[8] = {wih, whh, ain, aout, m0, m1, eaw, emw};
  _Float16*    dst[8] = {fwih, fwhh, fain, faout, fm0, fm1, feaw, femw};
  const int total = 204800;
  for (int u = blockIdx.x * blockDim.x + threadIdx.x; u < total; u += gridDim.x * blockDim.x) {
    int seg = 0, base = 0;
    while (seg < 7 && u >= base + seg_sz[seg]) { base += seg_sz[seg]; ++seg; }
    int idx  = u - base;
    int j    = idx & 15;
    int lane = (idx >> 4) & 31;
    int tile = idx >> 9;
    int nt   = tile % seg_nt[seg];
    int kc   = tile / seg_nt[seg];
    int ksrc = seg_k[seg];
    int n = nt * 16 + (lane & 15);
    int k = kc * 32 + (lane >> 4) * 8 + (j & 7) + (j >> 3) * 16;
    float v = (k < ksrc) ? src[seg][n * ksrc + k] : 0.f;
    dst[seg][idx] = (_Float16)v;
  }
}

// ---------------------------------------------------------------------------
// Kernel 1: aircraft encoder (WMMA, padded K) + sequential GRU + residual.
// 4 waves/WG, each wave owns 16 batch rows across all 64 steps.
// Hidden state lives in ping-pong f16 LDS tiles; gates processed one 16-col
// tile at a time in rolled loops (no big register arrays -> no spills).
// ---------------------------------------------------------------------------
__global__ __launch_bounds__(128) void gru_kernel(
    const float* __restrict__ obs, const float* __restrict__ h0,
    const float* __restrict__ eab,
    const float* __restrict__ bih, const float* __restrict__ bhh,
    const _Float16* __restrict__ fwih, const _Float16* __restrict__ fwhh,
    const _Float16* __restrict__ feaw,
    float* __restrict__ air_feat, float* __restrict__ h_last) {
  __shared__ alignas(16) _Float16 hA[4][16][kH];
  __shared__ alignas(16) _Float16 hB[4][16][kH];
  __shared__ alignas(16) _Float16 aeT[4][16][kH];
  __shared__ alignas(16) _Float16 obsT[4][16][32];   // K-padded obs tile

  const int wave = threadIdx.x >> 5;
  const int lane = threadIdx.x & 31;
  const int lc   = lane & 15;
  const int mb   = (lane >> 4) * 8;        // C-layout row offset for this lane half
  const int row0 = blockIdx.x * 64 + wave * 16;

  // zero obs pad columns once (stores each step only touch cols 0..14)
  for (int idx = lane; idx < 16 * 32; idx += 32)
    obsT[wave][idx >> 5][idx & 31] = (_Float16)0.f;

  // init hidden state from h0 into hA (C-layout scatter)
#pragma unroll 1
  for (int nt = 0; nt < 8; ++nt) {
    int col = nt * 16 + lc;
#pragma unroll
    for (int i = 0; i < 8; ++i)
      hA[wave][mb + i][col] = (_Float16)h0[(size_t)(row0 + mb + i) * kH + col];
  }
  __syncthreads();

  _Float16* cur = &hA[wave][0][0];
  _Float16* nxt = &hB[wave][0][0];

  for (int t = 0; t < kS; ++t) {
    // stage obs rows (cols 0..14) as f16
    for (int idx = lane; idx < 16 * kFOD; idx += 32) {
      int r = idx / kFOD, f = idx - r * kFOD;
      obsT[wave][r][f] = (_Float16)obs[((size_t)(row0 + r) * kS + t) * kFOD + f];
    }
    __syncthreads();
    // aircraft encoder via WMMA (K padded 15->32): ae = obs @ eaw.T + eab
#pragma unroll 1
    for (int nt = 0; nt < 8; ++nt) {
      v8f acc = {0.f, 0.f, 0.f, 0.f, 0.f, 0.f, 0.f, 0.f};
      acc = WMMA(load_afrag(&obsT[wave][0][0], 32, 0, lane), load_bfrag(feaw, nt, lane), acc);
      int col = nt * 16 + lc;
      float b = eab[col];
#pragma unroll
      for (int i = 0; i < 8; ++i)
        aeT[wave][mb + i][col] = (_Float16)(acc[i] + b);
    }
    __syncthreads();
    // gates, one 16-column tile at a time
#pragma unroll 1
    for (int nt = 0; nt < 8; ++nt) {
      int col = nt * 16 + lc;
      // r gate
      v8f r = {0.f, 0.f, 0.f, 0.f, 0.f, 0.f, 0.f, 0.f};
#pragma unroll
      for (int kc = 0; kc < 4; ++kc)
        r = WMMA(load_afrag(&aeT[wave][0][0], kH, kc * 32, lane),
                 load_bfrag(fwih, kc * 24 + nt, lane), r);
#pragma unroll
      for (int kc = 0; kc < 4; ++kc)
        r = WMMA(load_afrag(cur, kH, kc * 32, lane),
                 load_bfrag(fwhh, kc * 24 + nt, lane), r);
      float bs_r = bih[col] + bhh[col];
      // z gate
      v8f z = {0.f, 0.f, 0.f, 0.f, 0.f, 0.f, 0.f, 0.f};
#pragma unroll
      for (int kc = 0; kc < 4; ++kc)
        z = WMMA(load_afrag(&aeT[wave][0][0], kH, kc * 32, lane),
                 load_bfrag(fwih, kc * 24 + 8 + nt, lane), z);
#pragma unroll
      for (int kc = 0; kc < 4; ++kc)
        z = WMMA(load_afrag(cur, kH, kc * 32, lane),
                 load_bfrag(fwhh, kc * 24 + 8 + nt, lane), z);
      float bs_z = bih[kH + col] + bhh[kH + col];
      // n gate (input and hidden parts kept separate)
      v8f ai = {0.f, 0.f, 0.f, 0.f, 0.f, 0.f, 0.f, 0.f};
#pragma unroll
      for (int kc = 0; kc < 4; ++kc)
        ai = WMMA(load_afrag(&aeT[wave][0][0], kH, kc * 32, lane),
                  load_bfrag(fwih, kc * 24 + 16 + nt, lane), ai);
      v8f ah = {0.f, 0.f, 0.f, 0.f, 0.f, 0.f, 0.f, 0.f};
#pragma unroll
      for (int kc = 0; kc < 4; ++kc)
        ah = WMMA(load_afrag(cur, kH, kc * 32, lane),
                  load_bfrag(fwhh, kc * 24 + 16 + nt, lane), ah);
      float bi_n = bih[2 * kH + col], bh_n = bhh[2 * kH + col];
#pragma unroll
      for (int i = 0; i < 8; ++i) {
        float rv = 1.f / (1.f + __expf(-(r[i] + bs_r)));
        float zv = 1.f / (1.f + __expf(-(z[i] + bs_z)));
        float nn = tanhf((ai[i] + bi_n) + rv * (ah[i] + bh_n));
        float hold = (float)cur[(mb + i) * kH + col];
        float hv = (1.f - zv) * nn + zv * hold;
        nxt[(mb + i) * kH + col] = (_Float16)hv;
        air_feat[((size_t)(row0 + mb + i) * kS + t) * kH + col] =
            hv + (float)aeT[wave][mb + i][col];
      }
    }
    __syncthreads();
    _Float16* tmp = cur; cur = nxt; nxt = tmp;
  }
  // final hidden state -> d_out[N:]
#pragma unroll 1
  for (int nt = 0; nt < 8; ++nt) {
    int col = nt * 16 + lc;
#pragma unroll
    for (int i = 0; i < 8; ++i)
      h_last[(size_t)(row0 + mb + i) * kH + col] = (float)cur[(mb + i) * kH + col];
  }
}

// ---------------------------------------------------------------------------
// Kernel 2: missile encoders (WMMA, padded K) + LayerNorms + 4-head
// cross-attn + MLP head. 2 waves/WG, one 16-token tile per wave.
// ---------------------------------------------------------------------------
__global__ __launch_bounds__(64) void attn_mlp_kernel(
    const float* __restrict__ obs, const float* __restrict__ air_feat,
    const float* __restrict__ emb,
    const float* __restrict__ qlw, const float* __restrict__ qlb,
    const float* __restrict__ klw, const float* __restrict__ klb,
    const float* __restrict__ ainb, const float* __restrict__ aoutb,
    const float* __restrict__ m0b, const float* __restrict__ m1b,
    const float* __restrict__ fcow, const float* __restrict__ fcob,
    const _Float16* __restrict__ fain, const _Float16* __restrict__ faout,
    const _Float16* __restrict__ fm0, const _Float16* __restrict__ fm1,
    const _Float16* __restrict__ femw,
    float* __restrict__ val) {
  __shared__ alignas(16) _Float16 qn[2][16][kH];
  __shared__ alignas(16) _Float16 kn[2][2][16][kH];
  __shared__ alignas(16) _Float16 kr[2][2][16][kH];
  __shared__ alignas(16) _Float16 stA[2][16][kH];
  __shared__ alignas(16) _Float16 stB[2][16][kH];
  __shared__ alignas(16) _Float16 mrT[2][2][16][32];  // K-padded missile features
  __shared__ float mk[2][2][16];

  const int wave = threadIdx.x >> 5;
  const int lane = threadIdx.x & 31;
  const int lc   = lane & 15;
  const int mb   = (lane >> 4) * 8;
  const size_t tok0 = ((size_t)blockIdx.x * 2 + wave) * 16;

  // stage missile features (zero-padded) + inactive mask; lane -> row=lc, missile=lane>>4
  {
    int r = lc, m = lane >> 4;
#pragma unroll
    for (int c = 4; c < 32; ++c) mrT[wave][m][r][c] = (_Float16)0.f;
    bool msk = true;
#pragma unroll
    for (int f = 0; f < 4; ++f) {
      float v = obs[(tok0 + r) * kFOD + m * 4 + f];
      mrT[wave][m][r][f] = (_Float16)v;
      float tgt = (f & 1) ? 0.f : 1.f;
      float tol = 1e-8f + 1e-5f * tgt;     // isclose(atol=1e-8, rtol=1e-5)
      msk = msk && (fabsf(v - tgt) <= tol);
    }
    mk[wave][m][r] = msk ? 1.f : 0.f;
  }
  // query layernorm (lane -> row=lane>>1, half of 128 cols)
  {
    int r  = lane >> 1;
    int c0 = (lane & 1) * 64;
    float x[64];
    float s = 0.f, s2 = 0.f;
#pragma unroll
    for (int j = 0; j < 64; ++j) {
      float v = air_feat[(tok0 + r) * kH + c0 + j];
      x[j] = v; s += v; s2 += v * v;
    }
    s  += __shfl_xor(s, 1);
    s2 += __shfl_xor(s2, 1);
    float mu  = s * (1.f / 128.f);
    float var = s2 * (1.f / 128.f) - mu * mu;
    float rs  = rsqrtf(var + 1e-5f);
#pragma unroll
    for (int j = 0; j < 64; ++j) {
      int c = c0 + j;
      qn[wave][r][c] = (_Float16)((x[j] - mu) * rs * qlw[c] + qlb[c]);
    }
  }
  __syncthreads();

  // missile encoder via WMMA (K padded 4->32) + key layernorm
#pragma unroll 1
  for (int m = 0; m < 2; ++m) {
    float s[8]  = {0.f, 0.f, 0.f, 0.f, 0.f, 0.f, 0.f, 0.f};
    float s2[8] = {0.f, 0.f, 0.f, 0.f, 0.f, 0.f, 0.f, 0.f};
#pragma unroll 1
    for (int nt = 0; nt < 8; ++nt) {
      v8f acc = {0.f, 0.f, 0.f, 0.f, 0.f, 0.f, 0.f, 0.f};
      acc = WMMA(load_afrag(&mrT[wave][m][0][0], 32, 0, lane), load_bfrag(femw, nt, lane), acc);
      int col = nt * 16 + lc;
      float b = emb[col];
#pragma unroll
      for (int i = 0; i < 8; ++i) {
        float v = acc[i] + b;
        kr[wave][m][mb + i][col] = (_Float16)v;
        s[i] += v; s2[i] += v * v;
      }
    }
#pragma unroll
    for (int mw = 1; mw < 16; mw <<= 1) {
#pragma unroll
      for (int i = 0; i < 8; ++i) { s[i] += __shfl_xor(s[i], mw); s2[i] += __shfl_xor(s2[i], mw); }
    }
#pragma unroll 1
    for (int nt = 0; nt < 8; ++nt) {
      int col = nt * 16 + lc;
      float w = klw[col], bb = klb[col];
#pragma unroll
      for (int i = 0; i < 8; ++i) {
        float mu  = s[i] * (1.f / 128.f);
        float var = s2[i] * (1.f / 128.f) - mu * mu;
        float rs  = rsqrtf(var + 1e-5f);
        float v = (float)kr[wave][m][mb + i][col];
        kn[wave][m][mb + i][col] = (_Float16)((v - mu) * rs * w + bb);
      }
    }
  }
  __syncthreads();

  // per-head: q/k projections, scores, softmax, value projections, context
  const float scale = 0.1767766952966369f;  // 1/sqrt(32)
#pragma unroll 1
  for (int h4 = 0; h4 < 4; ++h4) {
    int ta = 2 * h4, tb = 2 * h4 + 1;
    int ca = ta * 16 + lc, cb = tb * 16 + lc;
    v8f q0 = {0.f, 0.f, 0.f, 0.f, 0.f, 0.f, 0.f, 0.f};
    v8f q1 = q0, k0a = q0, k0b = q0, k1a = q0, k1b = q0;
#pragma unroll
    for (int kc = 0; kc < 4; ++kc) {
      v16h aq = load_afrag(&qn[wave][0][0],    kH, kc * 32, lane);
      v16h a0 = load_afrag(&kn[wave][0][0][0], kH, kc * 32, lane);
      v16h a1 = load_afrag(&kn[wave][1][0][0], kH, kc * 32, lane);
      q0  = WMMA(aq, load_bfrag(fain, kc * 24 + ta, lane), q0);
      q1  = WMMA(aq, load_bfrag(fain, kc * 24 + tb, lane), q1);
      k0a = WMMA(a0, load_bfrag(fain, kc * 24 + 8 + ta, lane), k0a);
      k0b = WMMA(a0, load_bfrag(fain, kc * 24 + 8 + tb, lane), k0b);
      k1a = WMMA(a1, load_bfrag(fain, kc * 24 + 8 + ta, lane), k1a);
      k1b = WMMA(a1, load_bfrag(fain, kc * 24 + 8 + tb, lane), k1b);
    }
    float bqa = ainb[ca], bqb = ainb[cb];
    float bka = ainb[kH + ca], bkb = ainb[kH + cb];
    float p0[8], p1[8];
#pragma unroll
    for (int i = 0; i < 8; ++i) {
      float qa = q0[i] + bqa, qb = q1[i] + bqb;
      p0[i] = qa * (k0a[i] + bka) + qb * (k0b[i] + bkb);
      p1[i] = qa * (k1a[i] + bka) + qb * (k1b[i] + bkb);
    }
#pragma unroll
    for (int mw = 1; mw < 16; mw <<= 1) {
#pragma unroll
      for (int i = 0; i < 8; ++i) { p0[i] += __shfl_xor(p0[i], mw); p1[i] += __shfl_xor(p1[i], mw); }
    }
    float w0[8], w1[8];
#pragma unroll
    for (int i = 0; i < 8; ++i) {
      bool msk0 = mk[wave][0][mb + i] > 0.5f;
      bool msk1 = mk[wave][1][mb + i] > 0.5f;
      float s0 = p0[i] * scale, s1 = p1[i] * scale;
      if (msk0 && msk1)      { w0[i] = 0.f; w1[i] = 0.f; }
      else if (msk0)         { w0[i] = 0.f; w1[i] = 1.f; }
      else if (msk1)         { w0[i] = 1.f; w1[i] = 0.f; }
      else {
        float mx = fmaxf(s0, s1);
        float e0 = __expf(s0 - mx), e1 = __expf(s1 - mx);
        float inv = 1.f / (e0 + e1);
        w0[i] = e0 * inv; w1[i] = e1 * inv;
      }
    }
    // value projections for this head's two column tiles
    v8f v0a = {0.f, 0.f, 0.f, 0.f, 0.f, 0.f, 0.f, 0.f};
    v8f v0b = v0a, v1a = v0a, v1b = v0a;
#pragma unroll
    for (int kc = 0; kc < 4; ++kc) {
      v16h a0 = load_afrag(&kr[wave][0][0][0], kH, kc * 32, lane);
      v16h a1 = load_afrag(&kr[wave][1][0][0], kH, kc * 32, lane);
      v0a = WMMA(a0, load_bfrag(fain, kc * 24 + 16 + ta, lane), v0a);
      v0b = WMMA(a0, load_bfrag(fain, kc * 24 + 16 + tb, lane), v0b);
      v1a = WMMA(a1, load_bfrag(fain, kc * 24 + 16 + ta, lane), v1a);
      v1b = WMMA(a1, load_bfrag(fain, kc * 24 + 16 + tb, lane), v1b);
    }
    float bva = ainb[2 * kH + ca], bvb = ainb[2 * kH + cb];
#pragma unroll
    for (int i = 0; i < 8; ++i) {
      stA[wave][mb + i][ca] = (_Float16)(w0[i] * (v0a[i] + bva) + w1[i] * (v1a[i] + bva));
      stA[wave][mb + i][cb] = (_Float16)(w0[i] * (v0b[i] + bvb) + w1[i] * (v1b[i] + bvb));
    }
  }
  __syncthreads();
  // attention out projection + nan-guard + residual -> stB
#pragma unroll 1
  for (int nt = 0; nt < 8; ++nt) {
    v8f a = {0.f, 0.f, 0.f, 0.f, 0.f, 0.f, 0.f, 0.f};
#pragma unroll
    for (int kc = 0; kc < 4; ++kc)
      a = WMMA(load_afrag(&stA[wave][0][0], kH, kc * 32, lane),
               load_bfrag(faout, kc * 8 + nt, lane), a);
    int col = nt * 16 + lc;
    float b = aoutb[col];
#pragma unroll
    for (int i = 0; i < 8; ++i) {
      bool both = (mk[wave][0][mb + i] > 0.5f) && (mk[wave][1][mb + i] > 0.5f);
      float af = air_feat[(tok0 + mb + i) * kH + col];
      float ao = both ? 0.f : (a[i] + b);
      stB[wave][mb + i][col] = (_Float16)(af + ao);
    }
  }
  __syncthreads();
  // mlp0 + leaky relu -> stA
#pragma unroll 1
  for (int nt = 0; nt < 8; ++nt) {
    v8f a = {0.f, 0.f, 0.f, 0.f, 0.f, 0.f, 0.f, 0.f};
#pragma unroll
    for (int kc = 0; kc < 4; ++kc)
      a = WMMA(load_afrag(&stB[wave][0][0], kH, kc * 32, lane),
               load_bfrag(fm0, kc * 8 + nt, lane), a);
    int col = nt * 16 + lc;
    float b = m0b[col];
#pragma unroll
    for (int i = 0; i < 8; ++i) {
      float x = a[i] + b;
      x = (x >= 0.f) ? x : 0.01f * x;
      stA[wave][mb + i][col] = (_Float16)x;
    }
  }
  __syncthreads();
  // mlp1 + leaky relu + fco dot -> val
  float vp[8] = {0.f, 0.f, 0.f, 0.f, 0.f, 0.f, 0.f, 0.f};
#pragma unroll 1
  for (int nt = 0; nt < 8; ++nt) {
    v8f a = {0.f, 0.f, 0.f, 0.f, 0.f, 0.f, 0.f, 0.f};
#pragma unroll
    for (int kc = 0; kc < 4; ++kc)
      a = WMMA(load_afrag(&stA[wave][0][0], kH, kc * 32, lane),
               load_bfrag(fm1, kc * 8 + nt, lane), a);
    int col = nt * 16 + lc;
    float b = m1b[col];
    float w = fcow[col];
#pragma unroll
    for (int i = 0; i < 8; ++i) {
      float x = a[i] + b;
      x = (x >= 0.f) ? x : 0.01f * x;
      vp[i] += x * w;
    }
  }
#pragma unroll
  for (int mw = 1; mw < 16; mw <<= 1) {
#pragma unroll
    for (int i = 0; i < 8; ++i) vp[i] += __shfl_xor(vp[i], mw);
  }
  if (lc == 0) {
#pragma unroll
    for (int i = 0; i < 8; ++i) val[tok0 + mb + i] = vp[i] + fcob[0];
  }
}

// ---------------------------------------------------------------------------
extern "C" void kernel_launch(void* const* d_in, const int* in_sizes, int n_in,
                              void* d_out, int out_size, void* d_ws, size_t ws_size,
                              hipStream_t stream) {
  const float* obs   = (const float*)d_in[0];
  const float* h0    = (const float*)d_in[1];
  const float* eaw   = (const float*)d_in[2];
  const float* eab   = (const float*)d_in[3];
  const float* emw   = (const float*)d_in[4];
  const float* emb   = (const float*)d_in[5];
  const float* wih   = (const float*)d_in[6];
  const float* whh   = (const float*)d_in[7];
  const float* bih   = (const float*)d_in[8];
  const float* bhh   = (const float*)d_in[9];
  const float* qlw   = (const float*)d_in[10];
  const float* qlb   = (const float*)d_in[11];
  const float* klw   = (const float*)d_in[12];
  const float* klb   = (const float*)d_in[13];
  const float* ainw  = (const float*)d_in[14];
  const float* ainb  = (const float*)d_in[15];
  const float* aoutw = (const float*)d_in[16];
  const float* aoutb = (const float*)d_in[17];
  const float* m0w   = (const float*)d_in[18];
  const float* m0b   = (const float*)d_in[19];
  const float* m1w   = (const float*)d_in[20];
  const float* m1b   = (const float*)d_in[21];
  const float* fcow  = (const float*)d_in[22];
  const float* fcob  = (const float*)d_in[23];

  float* out    = (float*)d_out;
  float* val    = out;            // [N]
  float* h_last = out + kN;       // [B*H]

  char* ws = (char*)d_ws;
  float* air_feat = (float*)ws;                                   // N*128 f32
  size_t off = (size_t)kN * kH * sizeof(float);
  _Float16* fwih  = (_Float16*)(ws + off); off += 384 * 128 * 2;
  _Float16* fwhh  = (_Float16*)(ws + off); off += 384 * 128 * 2;
  _Float16* fain  = (_Float16*)(ws + off); off += 384 * 128 * 2;
  _Float16* faout = (_Float16*)(ws + off); off += 128 * 128 * 2;
  _Float16* fm0   = (_Float16*)(ws + off); off += 128 * 128 * 2;
  _Float16* fm1   = (_Float16*)(ws + off); off += 128 * 128 * 2;
  _Float16* feaw  = (_Float16*)(ws + off); off += 128 * 32 * 2;
  _Float16* femw  = (_Float16*)(ws + off); off += 128 * 32 * 2;

  prep_frags<<<dim3(800), dim3(256), 0, stream>>>(wih, whh, ainw, aoutw, m0w, m1w, eaw, emw,
                                                  fwih, fwhh, fain, faout, fm0, fm1, feaw, femw);
  gru_kernel<<<dim3(kB / 64), dim3(128), 0, stream>>>(obs, h0, eab, bih, bhh,
                                                      fwih, fwhh, feaw, air_feat, h_last);
  attn_mlp_kernel<<<dim3(kN / 32), dim3(64), 0, stream>>>(obs, air_feat, emb,
      qlw, qlb, klw, klb, ainb, aoutb, m0b, m1b, fcow, fcob,
      fain, faout, fm0, fm1, femw, val);
}
